// NAGNN_56126632624861
// MI455X (gfx1250) — matrix operand order repaired and checked by reading, not compiled, not run on hardware
//
#include <hip/hip_runtime.h>

// ---------------------------------------------------------------------------
// NAGNN forward for MI455X (gfx1250, wave32, WMMA bf16).
//   B=32 graphs, N=4096 nodes, E=16384 edges, F_IN=64, H=256, L=4
//   Pipeline per layer: memset(agg) -> scatter_add (f32 atomics, L2 resident)
//                       -> WMMA GEMM + fused bias/LayerNorm/ReLU -> bf16 x_l
//   Head: WMMA GEMM over concat(1088) + fused BatchNorm/ReLU/lin2-dot
//         -> atomicAdd logits -> masked softmax per graph.
//   Weights pre-converted to bf16; B tiles software-pipelined (double
//   buffered) so each v_wmma overlaps the next tile's b128 loads.
// ---------------------------------------------------------------------------

typedef __attribute__((ext_vector_type(16))) __bf16 v16bf;
typedef __attribute__((ext_vector_type(8)))  __bf16 v8bf;
typedef __attribute__((ext_vector_type(8)))  float  v8f;

#define BGR    32
#define NNODE  4096
#define NEDGE  16384
#define FIN    64
#define HDIM   256
#define NTOT   (BGR * NNODE)   /* 131072 */
#define MIDDIM 1088
#define OUTDIM 512

// ---------------------------------------------------------------------------
// Flat f32 -> bf16 conversion (for weight matrices; run once per call).
// ---------------------------------------------------------------------------
__global__ __launch_bounds__(256) void cvt_f32_bf16_kernel(
    const float* __restrict__ in, __bf16* __restrict__ out, int n)
{
  int i = blockIdx.x * blockDim.x + threadIdx.x;
  if (i < n) out[i] = (__bf16)in[i];
}

// ---------------------------------------------------------------------------
// Scatter-add: agg[dst] += x[src] over all edges. One thread per (edge, vec4).
// x array (<=134MB) fits in the 192MB L2, so gathers/atomics stay on-chip.
// ---------------------------------------------------------------------------
template <typename T>
__global__ __launch_bounds__(256) void scatter_add_kernel(
    const T* __restrict__ x, const int* __restrict__ links,
    float* __restrict__ agg, int d)
{
  long long tid = (long long)blockIdx.x * blockDim.x + threadIdx.x;
  const int vec = d >> 2;
  const long long total = (long long)BGR * NEDGE * vec;
  if (tid >= total) return;
  int c        = (int)(tid % vec);
  long long ed = tid / vec;
  int b = (int)(ed / NEDGE);
  int e = (int)(ed % NEDGE);
  const int* lb = links + (size_t)b * 2 * NEDGE;
  int src = lb[e]         + b * NNODE;
  int dst = lb[NEDGE + e] + b * NNODE;
  const T* xp = x   + (size_t)src * d + 4 * c;
  float*   ap = agg + (size_t)dst * d + 4 * c;
  atomicAdd(ap + 0, (float)xp[0]);
  atomicAdd(ap + 1, (float)xp[1]);
  atomicAdd(ap + 2, (float)xp[2]);
  atomicAdd(ap + 3, (float)xp[3]);
}

// ---------------------------------------------------------------------------
// GIN conv GEMM: h = agg[M,K] @ W[256,K]^T + b, then LayerNorm(256) + ReLU.
// One wave owns a 16(M) x 256(N) strip: 16 f32 accumulators, K-step of 32.
// A/B in contiguous k order (k = 16*half+e, same for both): A converted f32->bf16
// in-register (agg comes from f32 atomics); B loaded directly as bf16 with a
// one-deep software pipeline so WMMA t overlaps the loads for tile t+1.
// LN row stats via 4x shfl_xor inside each 16-lane half.
// ---------------------------------------------------------------------------
__global__ __launch_bounds__(256) void gin_gemm_ln_relu(
    const float* __restrict__ agg, const __bf16* __restrict__ Wb,
    const float* __restrict__ bias, const float* __restrict__ gamma,
    const float* __restrict__ beta, __bf16* __restrict__ xout, int K)
{
  const int lane = threadIdx.x & 31;
  const int wave = threadIdx.x >> 5;
  const int half = lane >> 4;
  const int l15  = lane & 15;
  const int m0   = (blockIdx.x * 8 + wave) * 16;

  v8f zero = {};
  v8f acc[16];
#pragma unroll
  for (int t = 0; t < 16; ++t) acc[t] = zero;

  const float* aptr = agg + (size_t)(m0 + l15) * K;

  for (int k0 = 0; k0 < K; k0 += 32) {
    // A tile 16x32: lane row = l15; elems e -> K = 16*half+e
    v16bf amat;
    const float* p0 = aptr + k0 + 16 * half;
#pragma unroll
    for (int e = 0; e < 8; ++e) amat[e]     = (__bf16)p0[e];
#pragma unroll
    for (int e = 0; e < 8; ++e) amat[8 + e] = (__bf16)p0[8 + e];

    // B tile 32x16: lane col = l15; elems e -> K = 16*half+e; B[k][n]=W[n][k].
    // 32 contiguous bf16 bytes, 32B-aligned -> 2x b128, no VALU.
    // Software pipeline depth 1: load tile t+1 before WMMA of tile t.
    const __bf16* wbase = Wb + (size_t)l15 * K + k0 + 16 * half;
    v16bf bcur = *reinterpret_cast<const v16bf*>(wbase);
#pragma unroll
    for (int t = 0; t < 15; ++t) {
      const v16bf bnext = *reinterpret_cast<const v16bf*>(
          wbase + (size_t)(t + 1) * 16 * K);
      acc[t] = __builtin_amdgcn_wmma_f32_16x16x32_bf16(
          false, amat, false, bcur, (short)0, acc[t], false, false);
      bcur = bnext;
    }
    acc[15] = __builtin_amdgcn_wmma_f32_16x16x32_bf16(
        false, amat, false, bcur, (short)0, acc[15], false, false);
  }

  // Epilogue: bias add, LayerNorm over 256 cols per row, ReLU, store bf16.
  float g[16], be[16];
#pragma unroll
  for (int t = 0; t < 16; ++t) {
    const int n = 16 * t + l15;
    const float bb = bias[n];
    g[t]  = gamma[n];
    be[t] = beta[n];
#pragma unroll
    for (int r = 0; r < 8; ++r) acc[t][r] += bb;
  }

#pragma unroll
  for (int r = 0; r < 8; ++r) {
    // row M = m0 + r + 8*half lives entirely in this lane's 16-lane half
    float s = 0.0f, s2 = 0.0f;
#pragma unroll
    for (int t = 0; t < 16; ++t) { float v = acc[t][r]; s += v; s2 += v * v; }
#pragma unroll
    for (int off = 1; off < 16; off <<= 1) {
      s  += __shfl_xor(s,  off, 32);
      s2 += __shfl_xor(s2, off, 32);
    }
    const float mean = s  * (1.0f / 256.0f);
    const float var  = s2 * (1.0f / 256.0f) - mean * mean;
    const float rstd = rsqrtf(var + 1e-5f);
    const size_t row = (size_t)(m0 + r + 8 * half) * HDIM;
#pragma unroll
    for (int t = 0; t < 16; ++t) {
      float v = (acc[t][r] - mean) * rstd * g[t] + be[t];
      v = fmaxf(v, 0.0f);
      xout[row + 16 * t + l15] = (__bf16)v;
    }
  }
}

// ---------------------------------------------------------------------------
// Head: h = concat(x0,x1..x4)[M,1088] @ W1[512,1088]^T, BN-affine, ReLU, then
// logits[m] += h[m,:] . W2  (lin2 fused as per-row dot + cross-lane reduce +
// atomicAdd). A-source selected per K-block from the 5 layer outputs, so the
// 1088-wide concat tensor is never materialized.
// ---------------------------------------------------------------------------
__global__ __launch_bounds__(256) void head_gemm_kernel(
    const float*  __restrict__ x0,
    const __bf16* __restrict__ x1, const __bf16* __restrict__ x2,
    const __bf16* __restrict__ x3, const __bf16* __restrict__ x4,
    const __bf16* __restrict__ W1b, const float* __restrict__ b1,
    const float* __restrict__ bnG, const float* __restrict__ bnB,
    const float* __restrict__ bnM, const float* __restrict__ bnV,
    const float* __restrict__ W2,
    float* __restrict__ logits)
{
  const int lane = threadIdx.x & 31;
  const int wave = threadIdx.x >> 5;
  const int half = lane >> 4;
  const int l15  = lane & 15;
  const int m0     = (blockIdx.x * 8 + wave) * 16;
  const int n_base = blockIdx.y * 128;
  const __bf16* xs[4] = {x1, x2, x3, x4};

  v8f zero = {};
  v8f acc[8];
#pragma unroll
  for (int t = 0; t < 8; ++t) acc[t] = zero;

  const int arow = m0 + l15;

  for (int k0 = 0; k0 < MIDDIM; k0 += 32) {
    v16bf amat;
    if (k0 < FIN) {
      const float* p = x0 + (size_t)arow * FIN + k0 + 16 * half;
#pragma unroll
      for (int e = 0; e < 8; ++e) amat[e]     = (__bf16)p[e];
#pragma unroll
      for (int e = 0; e < 8; ++e) amat[8 + e] = (__bf16)p[8 + e];
    } else {
      const int li = (k0 - FIN) >> 8;
      const int kl = (k0 - FIN) & 255;
      const __bf16* p = xs[li] + (size_t)arow * HDIM + kl + 16 * half;
      const v8bf lo = *reinterpret_cast<const v8bf*>(p);       // 16B aligned
      const v8bf hi = *reinterpret_cast<const v8bf*>(p + 8);   // 16B aligned
#pragma unroll
      for (int e = 0; e < 8; ++e) { amat[e] = lo[e]; amat[8 + e] = hi[e]; }
    }
    // Software pipeline depth 1 on the B tiles.
    const __bf16* wbase = W1b + (size_t)(n_base + l15) * MIDDIM + k0 + 16 * half;
    v16bf bcur = *reinterpret_cast<const v16bf*>(wbase);
#pragma unroll
    for (int t = 0; t < 7; ++t) {
      const v16bf bnext = *reinterpret_cast<const v16bf*>(
          wbase + (size_t)(t + 1) * 16 * MIDDIM);
      acc[t] = __builtin_amdgcn_wmma_f32_16x16x32_bf16(
          false, amat, false, bcur, (short)0, acc[t], false, false);
      bcur = bnext;
    }
    acc[7] = __builtin_amdgcn_wmma_f32_16x16x32_bf16(
        false, amat, false, bcur, (short)0, acc[7], false, false);
  }

  // Fused epilogue: BatchNorm (eval) affine + ReLU + dot with lin2 weights.
  float scale[8], shift[8], bia[8], w2[8];
#pragma unroll
  for (int t = 0; t < 8; ++t) {
    const int n = n_base + 16 * t + l15;
    const float sc = bnG[n] * rsqrtf(bnV[n] + 1e-5f);
    scale[t] = sc;
    shift[t] = bnB[n] - bnM[n] * sc;
    bia[t]   = b1[n];
    w2[t]    = W2[n];
  }

#pragma unroll
  for (int r = 0; r < 8; ++r) {
    float part = 0.0f;
#pragma unroll
    for (int t = 0; t < 8; ++t) {
      float v = (acc[t][r] + bia[t]) * scale[t] + shift[t];
      v = fmaxf(v, 0.0f);
      part += v * w2[t];
    }
#pragma unroll
    for (int off = 1; off < 16; off <<= 1) part += __shfl_xor(part, off, 32);
    if (l15 == 0) atomicAdd(&logits[m0 + r + 8 * half], part);
  }
}

__global__ void init_logits_kernel(float* __restrict__ logits,
                                   const float* __restrict__ b2)
{
  int i = blockIdx.x * blockDim.x + threadIdx.x;
  if (i < NTOT) logits[i] = b2[0];
}

// Masked softmax over 4096 nodes per graph; one block per graph.
__global__ __launch_bounds__(256) void softmax_kernel(
    const float* __restrict__ logits, const unsigned char* __restrict__ mask,
    float* __restrict__ out)
{
  const int b = blockIdx.x;
  const float* lg = logits + (size_t)b * NNODE;
  const unsigned char* mk = mask + (size_t)b * NNODE;
  float* op = out + (size_t)b * NNODE;

  __shared__ float red[8];
  float vals[16];
  float lmax = -3.0e38f;
#pragma unroll
  for (int j = 0; j < 16; ++j) {
    const int i = threadIdx.x + 256 * j;
    const float v = mk[i] ? lg[i] : -1.0e8f;
    vals[j] = v;
    lmax = fmaxf(lmax, v);
  }
#pragma unroll
  for (int off = 1; off < 32; off <<= 1)
    lmax = fmaxf(lmax, __shfl_xor(lmax, off, 32));
  const int wv = threadIdx.x >> 5;
  const int ln = threadIdx.x & 31;
  if (ln == 0) red[wv] = lmax;
  __syncthreads();
  float gmax = red[0];
#pragma unroll
  for (int w = 1; w < 8; ++w) gmax = fmaxf(gmax, red[w]);
  __syncthreads();

  float lsum = 0.0f;
#pragma unroll
  for (int j = 0; j < 16; ++j) { vals[j] = __expf(vals[j] - gmax); lsum += vals[j]; }
#pragma unroll
  for (int off = 1; off < 32; off <<= 1) lsum += __shfl_xor(lsum, off, 32);
  if (ln == 0) red[wv] = lsum;
  __syncthreads();
  float gsum = 0.0f;
#pragma unroll
  for (int w = 0; w < 8; ++w) gsum += red[w];
  const float inv = 1.0f / gsum;
#pragma unroll
  for (int j = 0; j < 16; ++j) op[threadIdx.x + 256 * j] = vals[j] * inv;
}

// ---------------------------------------------------------------------------
extern "C" void kernel_launch(void* const* d_in, const int* in_sizes, int n_in,
                              void* d_out, int out_size, void* d_ws, size_t ws_size,
                              hipStream_t stream)
{
  (void)in_sizes; (void)out_size; (void)ws_size;

  const float*         nodes = (const float*)d_in[0];
  const int*           links = (const int*)d_in[1];
  const unsigned char* mask  = (const unsigned char*)d_in[2];

  const float *convW[4], *convB[4], *lnG[4], *lnB[4];
  const float *W1, *b1, *bnG, *bnB, *bnM, *bnV, *W2, *b2;
  if (n_in >= 27) {
    // Python lists expanded into individual d_in slots.
    for (int l = 0; l < 4; ++l) {
      convW[l] = (const float*)d_in[3 + l];
      convB[l] = (const float*)d_in[7 + l];
      lnG[l]   = (const float*)d_in[11 + l];
      lnB[l]   = (const float*)d_in[15 + l];
    }
    W1  = (const float*)d_in[19]; b1  = (const float*)d_in[20];
    bnG = (const float*)d_in[21]; bnB = (const float*)d_in[22];
    bnM = (const float*)d_in[23]; bnV = (const float*)d_in[24];
    W2  = (const float*)d_in[25]; b2  = (const float*)d_in[26];
  } else {
    // Python lists concatenated flat.
    const float* cw = (const float*)d_in[3];
    convW[0] = cw; cw += FIN * HDIM;
    for (int l = 1; l < 4; ++l) { convW[l] = cw; cw += HDIM * HDIM; }
    for (int l = 0; l < 4; ++l) {
      convB[l] = (const float*)d_in[4] + l * HDIM;
      lnG[l]   = (const float*)d_in[5] + l * HDIM;
      lnB[l]   = (const float*)d_in[6] + l * HDIM;
    }
    W1  = (const float*)d_in[7];  b1  = (const float*)d_in[8];
    bnG = (const float*)d_in[9];  bnB = (const float*)d_in[10];
    bnM = (const float*)d_in[11]; bnV = (const float*)d_in[12];
    W2  = (const float*)d_in[13]; b2  = (const float*)d_in[14];
  }

  // Workspace carve-up: agg f32 [NT,256]; x1..x4 bf16 [NT,256]; logits f32 [NT];
  // bf16 weight copies (convW x4, W1).
  char* ws = (char*)d_ws;
  float* agg = (float*)ws;  ws += (size_t)NTOT * HDIM * sizeof(float);
  __bf16* xl[4];
  for (int l = 0; l < 4; ++l) { xl[l] = (__bf16*)ws; ws += (size_t)NTOT * HDIM * sizeof(__bf16); }
  float* logits = (float*)ws; ws += (size_t)NTOT * sizeof(float);
  __bf16* convWb[4];
  convWb[0] = (__bf16*)ws;  ws += (size_t)FIN * HDIM * sizeof(__bf16);
  for (int l = 1; l < 4; ++l) { convWb[l] = (__bf16*)ws; ws += (size_t)HDIM * HDIM * sizeof(__bf16); }
  __bf16* W1b = (__bf16*)ws;

  // ---- Pre-convert weights to bf16 (tiny; keeps WMMA loop free of v_cvt) ----
  {
    int n = FIN * HDIM;
    cvt_f32_bf16_kernel<<<(n + 255) / 256, 256, 0, stream>>>(convW[0], convWb[0], n);
    n = HDIM * HDIM;
    for (int l = 1; l < 4; ++l)
      cvt_f32_bf16_kernel<<<(n + 255) / 256, 256, 0, stream>>>(convW[l], convWb[l], n);
    n = OUTDIM * MIDDIM;
    cvt_f32_bf16_kernel<<<(n + 255) / 256, 256, 0, stream>>>(W1, W1b, n);
  }

  // ---- Layer 1 (K = 64, f32 source) ----
  hipMemsetAsync(agg, 0, (size_t)NTOT * FIN * sizeof(float), stream);
  {
    const long long total = (long long)BGR * NEDGE * (FIN / 4);
    const int blocks = (int)((total + 255) / 256);
    scatter_add_kernel<float><<<blocks, 256, 0, stream>>>(nodes, links, agg, FIN);
  }
  gin_gemm_ln_relu<<<NTOT / 128, 256, 0, stream>>>(
      agg, convWb[0], convB[0], lnG[0], lnB[0], xl[0], FIN);

  // ---- Layers 2..4 (K = 256, bf16 source) ----
  for (int l = 1; l < 4; ++l) {
    hipMemsetAsync(agg, 0, (size_t)NTOT * HDIM * sizeof(float), stream);
    const long long total = (long long)BGR * NEDGE * (HDIM / 4);
    const int blocks = (int)((total + 255) / 256);
    scatter_add_kernel<__bf16><<<blocks, 256, 0, stream>>>(xl[l - 1], links, agg, HDIM);
    gin_gemm_ln_relu<<<NTOT / 128, 256, 0, stream>>>(
        agg, convWb[l], convB[l], lnG[l], lnB[l], xl[l], HDIM);
  }

  // ---- Head: lin1 + BN + ReLU + lin2 fused, then masked softmax ----
  init_logits_kernel<<<NTOT / 256, 256, 0, stream>>>(logits, b2);
  dim3 hgrid(NTOT / 128, OUTDIM / 128);
  head_gemm_kernel<<<hgrid, 256, 0, stream>>>(
      nodes, xl[0], xl[1], xl[2], xl[3],
      W1b, b1, bnG, bnB, bnM, bnV, W2, logits);
  softmax_kernel<<<BGR, 256, 0, stream>>>(logits, mask, (float*)d_out);
}
